// RationalFourierAttention_40853728919815
// MI455X (gfx1250) — matrix operand order repaired and checked by reading, not compiled, not run on hardware
//
#include <hip/hip_runtime.h>
#include <stdint.h>

// ---------------------------------------------------------------------------
// RationalFourierAttention for MI455X (gfx1250, wave32, WMMA bf16 16x16x32)
// B=2 S=2048 D=1024 H=16 DH=64 F=32 (2F=64). NORM = 1/sqrt(64) = 0.125
//
// Softmax stability: qf/kf entries = {cos,sin}(g)*NORM with g in (0,1], hence
// in (0, 0.125]; scores <= temp (= 1). exp of bounded-small scores cannot
// overflow -> softmax without running-max is exact: no cross-lane reductions
// or O-rescale in the inner loop. log2(e)*temperature is folded into qf, so
// the inner-loop exponential is a bare v_exp_f32 (exp2). P-tile bf16 uses
// truncation (high 16 bits) so the store is a single b16 op per value.
// ---------------------------------------------------------------------------

#define S_LEN   2048
#define DMODEL  1024
#define NHEAD   16
#define DHEAD   64
#define MROWS   4096      // B*S
#define NORM_F  0.125f
#define LOG2E_F 1.44269504088896340736f

typedef __attribute__((ext_vector_type(16))) __bf16 v16bf;
typedef __attribute__((ext_vector_type(8)))  float  v8f;

__device__ __forceinline__ uint16_t f32_to_bf16(float x) {
  union { float f; uint32_t u; } c; c.f = x;
  uint32_t r = c.u + 0x7FFFu + ((c.u >> 16) & 1u);   // round-to-nearest-even
  return (uint16_t)(r >> 16);
}
__device__ __forceinline__ uint16_t f32_to_bf16_trunc(float x) {
  union { float f; uint32_t u; } c; c.f = x;
  return (uint16_t)(c.u >> 16);                      // truncate: 1 mov / d16_hi store
}
__device__ __forceinline__ float bf16_to_f32(uint16_t h) {
  union { uint32_t u; float f; } c; c.u = ((uint32_t)h) << 16;
  return c.f;
}

// Two 16B loads -> one 16-element bf16 WMMA fragment register set.
__device__ __forceinline__ v16bf load_frag_2x16B(const uint16_t* p0, const uint16_t* p1) {
  union { v16bf v; uint4 u[2]; } f;
  f.u[0] = *(const uint4*)p0;
  f.u[1] = *(const uint4*)p1;
  return f.v;
}

// 16-lane group sum (wave32: lanes 0-15 hold rows j, lanes 16-31 rows j+8).
// Used ONCE per row at the very end (not in the inner loop).
__device__ __forceinline__ float redsum16(float x) {
  #pragma unroll
  for (int o = 1; o < 16; o <<= 1) x += __shfl_xor(x, o, 16);
  return x;
}

// ---------------------------------------------------------------------------
// f32 -> bf16 elementwise convert
// ---------------------------------------------------------------------------
__global__ void cvt_f32_bf16(const float* __restrict__ in, uint16_t* __restrict__ out, int n) {
  int i = blockIdx.x * 256 + threadIdx.x;
  if (i < n) out[i] = f32_to_bf16(in[i]);
}

// ---------------------------------------------------------------------------
// C = A[4096x1024] @ W^T, W is [1024x1024] row-major (e,d) -> B-col e is W row e.
// Wave computes 16x64 C tile (4 wmma accs, A frag reused). WG = 8 waves = 128x64.
// MODE 0: write bf16 head-split [B,H,S,DH]          (Q, K pre-feature)
// MODE 1: write bf16 transposed [B,H,DH,S]          (V for the PV GEMM)
// MODE 2: write f32 plain       [M, N]              (final output projection)
// Weights (2MB bf16) + activations stay resident in 192MB L2 -> direct loads.
// ---------------------------------------------------------------------------
template<int MODE>
__global__ __launch_bounds__(256) void gemm_bf16(const uint16_t* __restrict__ A,
                                                 const uint16_t* __restrict__ W,
                                                 float* __restrict__ Cf,
                                                 uint16_t* __restrict__ Cb) {
  const int K = DMODEL;
  const int wave = threadIdx.x >> 5;
  const int lane = threadIdx.x & 31;
  const int half = lane >> 4;        // K-half select per ISA fragment layout
  const int lm   = lane & 15;        // row (A) / col (B,C)
  const int n0 = blockIdx.x * 64;
  const int mt = blockIdx.y * 128 + wave * 16;

  v8f acc[4];
  #pragma unroll
  for (int t = 0; t < 4; ++t)
    #pragma unroll
    for (int j = 0; j < 8; ++j) acc[t][j] = 0.0f;

  const uint16_t* arow = A + (size_t)(mt + lm) * K;
  const int aoff = half * 8;    // A frag: lanes0-15 K{0..7,16..23}, lanes16-31 K{8..15,24..31}
  const int boff = half * 8;    // B frag: lanes0-15 K{0..7,16..23}, lanes16-31 K{8..15,24..31}

  for (int k0 = 0; k0 < K; k0 += 32) {
    v16bf af = load_frag_2x16B(arow + k0 + aoff, arow + k0 + aoff + 16);
    #pragma unroll
    for (int t = 0; t < 4; ++t) {
      const uint16_t* brow = W + (size_t)(n0 + t * 16 + lm) * K + k0 + boff;
      v16bf bfr = load_frag_2x16B(brow, brow + 16);
      acc[t] = __builtin_amdgcn_wmma_f32_16x16x32_bf16(false, af, false, bfr,
                                                       (short)0, acc[t], false, false);
    }
  }

  #pragma unroll
  for (int t = 0; t < 4; ++t) {
    const int e = n0 + t * 16 + lm;
    #pragma unroll
    for (int j = 0; j < 8; ++j) {
      const int m = mt + j + 8 * half;           // C layout: VGPR j holds rows j / j+8
      const float v = acc[t][j];
      if (MODE == 0) {
        const int b = m >> 11, s = m & 2047, h = e >> 6, dh = e & 63;
        Cb[((((size_t)b * NHEAD + h) * S_LEN + s) << 6) + dh] = f32_to_bf16(v);
      } else if (MODE == 1) {
        const int b = m >> 11, s = m & 2047, h = e >> 6, dh = e & 63;
        Cb[(((size_t)b * NHEAD + h) * DHEAD + dh) * S_LEN + s] = f32_to_bf16(v);
      } else {
        Cf[(size_t)m * DMODEL + e] = v;
      }
    }
  }
}

// ---------------------------------------------------------------------------
// Random-Fourier features: per row (b,h,s): proj_f = q.freq_f + phase[f,0];
// g = exp(-sigma^2 proj^2); out = [cos(g), sin(g)] * scale, with
//   scale = NORM * temperature * log2(e)   on the Q side (QSIDE=1)
//   scale = NORM                           on the K side
// One wave per row; lane f owns feature f. q loads are wave-uniform (scalar).
// ---------------------------------------------------------------------------
template<int QSIDE>
__global__ __launch_bounds__(256) void rf_feat(const uint16_t* __restrict__ qhead,
                                               const float* __restrict__ freq,
                                               const float* __restrict__ phase,
                                               const float* __restrict__ sigma,
                                               const float* __restrict__ temperature,
                                               uint16_t* __restrict__ outf) {
  const int row = blockIdx.x * 8 + (threadIdx.x >> 5);
  const int f   = threadIdx.x & 31;
  const uint16_t* q  = qhead + (size_t)row * DHEAD;
  const float*    fr = freq + f * DHEAD;
  float p = 0.0f;
  #pragma unroll 16
  for (int d = 0; d < DHEAD; ++d) p = fmaf(bf16_to_f32(q[d]), fr[d], p);
  p += phase[f * DHEAD];                 // phase[:,0] per-feature phase
  const float sg = sigma[0];
  const float g  = __expf(-(sg * sg) * p * p);
  float scale = NORM_F;
  if (QSIDE) scale *= temperature[0] * LOG2E_F;
  outf[(size_t)row * 64 + f]      = f32_to_bf16(__cosf(g) * scale);
  outf[(size_t)row * 64 + 32 + f] = f32_to_bf16(__sinf(g) * scale);
}

// ---------------------------------------------------------------------------
// Streaming softmax attention (qf,kf: [BH,S,64] bf16; vt: [BH,64,S] bf16).
// WG = 4 waves; each wave owns TWO 16-row query tiles (32 q rows), so the kf
// and V fragments loaded per 32-key step feed 16 WMMAs. Scores are bounded
// and pre-scaled by log2e -> p = exp2(s) (bare v_exp_f32), O accumulates
// unscaled, row sums kept as per-lane partials (f32), reduced once at the
// end. P tiles transit per-wave LDS (C-layout -> A-fragment layout transpose,
// intra-wave s_wait_dscnt). Out: bf16 [B,S,D] (D index = h*64+dh).
// ---------------------------------------------------------------------------
__global__ __launch_bounds__(128) void rfattn(const uint16_t* __restrict__ qf,
                                              const uint16_t* __restrict__ kf,
                                              const uint16_t* __restrict__ vt,
                                              uint16_t* __restrict__ outa) {
  __shared__ uint16_t ptile[4][2][16 * 32];   // per-wave, per-qtile 16x32 bf16 P
  const int wave = threadIdx.x >> 5;
  const int lane = threadIdx.x & 31;
  const int half = lane >> 4;
  const int lm   = lane & 15;
  const int bh = blockIdx.y;
  const int b = bh >> 4, h = bh & 15;
  const int q0 = blockIdx.x * 128 + wave * 32;

  const uint16_t* qbase = qf + (size_t)bh * S_LEN * 64;
  const uint16_t* kbase = kf + (size_t)bh * S_LEN * 64;
  const uint16_t* vbase = vt + (size_t)bh * DHEAD * S_LEN;

  // Persistent Q fragments: [qtile][feature-chunk]
  v16bf aq[2][2];
  #pragma unroll
  for (int qt = 0; qt < 2; ++qt) {
    const uint16_t* qr = qbase + (size_t)(q0 + qt * 16 + lm) * 64 + half * 8;
    aq[qt][0] = load_frag_2x16B(qr,      qr + 16);
    aq[qt][1] = load_frag_2x16B(qr + 32, qr + 48);
  }

  v8f acc[2][4];
  float lacc[2][8];                       // per-lane partial row sums
  #pragma unroll
  for (int qt = 0; qt < 2; ++qt) {
    #pragma unroll
    for (int j = 0; j < 8; ++j) lacc[qt][j] = 0.0f;
    #pragma unroll
    for (int t = 0; t < 4; ++t)
      #pragma unroll
      for (int j = 0; j < 8; ++j) acc[qt][t][j] = 0.0f;
  }

  for (int kb = 0; kb < S_LEN; kb += 32) {
    // ---- kf B-fragments (shared by both q tiles) --------------------------
    const uint16_t* kr0 = kbase + (size_t)(kb + lm) * 64 + half * 8;
    const uint16_t* kr1 = kbase + (size_t)(kb + 16 + lm) * 64 + half * 8;
    v16bf b00 = load_frag_2x16B(kr0,      kr0 + 16);
    v16bf b01 = load_frag_2x16B(kr0 + 32, kr0 + 48);
    v16bf b10 = load_frag_2x16B(kr1,      kr1 + 16);
    v16bf b11 = load_frag_2x16B(kr1 + 32, kr1 + 48);

    // ---- scores + exp2 + P-to-LDS ----------------------------------------
    #pragma unroll
    for (int qt = 0; qt < 2; ++qt) {
      v8f s0, s1;
      #pragma unroll
      for (int j = 0; j < 8; ++j) { s0[j] = 0.0f; s1[j] = 0.0f; }
      s0 = __builtin_amdgcn_wmma_f32_16x16x32_bf16(false, aq[qt][0], false, b00, (short)0, s0, false, false);
      s0 = __builtin_amdgcn_wmma_f32_16x16x32_bf16(false, aq[qt][1], false, b01, (short)0, s0, false, false);
      s1 = __builtin_amdgcn_wmma_f32_16x16x32_bf16(false, aq[qt][0], false, b10, (short)0, s1, false, false);
      s1 = __builtin_amdgcn_wmma_f32_16x16x32_bf16(false, aq[qt][1], false, b11, (short)0, s1, false, false);
      uint16_t* pt = &ptile[wave][qt][0];
      #pragma unroll
      for (int j = 0; j < 8; ++j) {
        // log2e folded into qf -> exp(score) == exp2(s): bare v_exp_f32
        const float p0 = __builtin_amdgcn_exp2f(s0[j]);
        const float p1 = __builtin_amdgcn_exp2f(s1[j]);
        lacc[qt][j] += p0 + p1;
        const int r = j + 8 * half;       // C-layout row for this lane
        pt[r * 32 + lm]      = f32_to_bf16_trunc(p0);
        pt[r * 32 + 16 + lm] = f32_to_bf16_trunc(p1);
      }
    }
    // Intra-wave LDS RAW (cross-lane): drain DS before re-reading P tiles.
    asm volatile("s_wait_dscnt 0" ::: "memory");

    v16bf ap[2];
    #pragma unroll
    for (int qt = 0; qt < 2; ++qt) {
      const uint16_t* pr = &ptile[wave][qt][0] + lm * 32 + half * 8;
      ap[qt] = load_frag_2x16B(pr, pr + 16);  // A-fragment layout read
    }

    // ---- O += P @ V (V fragments shared by both q tiles) ------------------
    #pragma unroll
    for (int t = 0; t < 4; ++t) {
      const uint16_t* vr = vbase + (size_t)(t * 16 + lm) * S_LEN + kb + half * 8;
      v16bf bv = load_frag_2x16B(vr, vr + 16);
      acc[0][t] = __builtin_amdgcn_wmma_f32_16x16x32_bf16(false, ap[0], false, bv,
                                                          (short)0, acc[0][t], false, false);
      acc[1][t] = __builtin_amdgcn_wmma_f32_16x16x32_bf16(false, ap[1], false, bv,
                                                          (short)0, acc[1][t], false, false);
    }
  }

  // ---- normalize (single cross-lane reduction) + store bf16 [B,S,D] -------
  #pragma unroll
  for (int qt = 0; qt < 2; ++qt) {
    float rinv[8];
    #pragma unroll
    for (int j = 0; j < 8; ++j) rinv[j] = 1.0f / redsum16(lacc[qt][j]);
    #pragma unroll
    for (int t = 0; t < 4; ++t) {
      const int e = h * 64 + t * 16 + lm;
      #pragma unroll
      for (int j = 0; j < 8; ++j) {
        const int s = q0 + qt * 16 + j + 8 * half;
        outa[((size_t)b * S_LEN + s) * DMODEL + e] = f32_to_bf16(acc[qt][t][j] * rinv[j]);
      }
    }
  }
}

// ---------------------------------------------------------------------------
// Launch
// ---------------------------------------------------------------------------
extern "C" void kernel_launch(void* const* d_in, const int* in_sizes, int n_in,
                              void* d_out, int out_size, void* d_ws, size_t ws_size,
                              hipStream_t stream) {
  (void)in_sizes; (void)n_in; (void)out_size; (void)ws_size;

  const float* x   = (const float*)d_in[0];
  const float* wq  = (const float*)d_in[1];
  const float* wk  = (const float*)d_in[2];
  const float* wv  = (const float*)d_in[3];
  const float* wo  = (const float*)d_in[4];
  const float* frq = (const float*)d_in[5];
  const float* phq = (const float*)d_in[6];
  const float* sgq = (const float*)d_in[7];
  const float* frk = (const float*)d_in[8];
  const float* phk = (const float*)d_in[9];
  const float* sgk = (const float*)d_in[10];
  const float* tmp = (const float*)d_in[11];

  char* ws = (char*)d_ws;
  uint16_t* xb  = (uint16_t*)(ws);                          // 8 MB  x bf16
  uint16_t* wqb = (uint16_t*)(ws + ((size_t) 8 << 20));     // 2 MB
  uint16_t* wkb = (uint16_t*)(ws + ((size_t)10 << 20));     // 2 MB
  uint16_t* wvb = (uint16_t*)(ws + ((size_t)12 << 20));     // 2 MB
  uint16_t* wob = (uint16_t*)(ws + ((size_t)14 << 20));     // 2 MB
  uint16_t* qh  = (uint16_t*)(ws + ((size_t)16 << 20));     // 8 MB  Q heads bf16
  uint16_t* kh  = (uint16_t*)(ws + ((size_t)24 << 20));     // 8 MB  K heads bf16
  uint16_t* vtb = (uint16_t*)(ws + ((size_t)32 << 20));     // 8 MB  V^T bf16
  uint16_t* qfb = (uint16_t*)(ws + ((size_t)40 << 20));     // 8 MB  qf bf16
  uint16_t* kfb = (uint16_t*)(ws + ((size_t)48 << 20));     // 8 MB  kf bf16
  uint16_t* oab = (uint16_t*)(ws + ((size_t)56 << 20));     // 8 MB  attn out bf16

  const int NX = MROWS * DMODEL;     // 4194304
  const int NW = DMODEL * DMODEL;    // 1048576
  cvt_f32_bf16<<<NX / 256, 256, 0, stream>>>(x,  xb,  NX);
  cvt_f32_bf16<<<NW / 256, 256, 0, stream>>>(wq, wqb, NW);
  cvt_f32_bf16<<<NW / 256, 256, 0, stream>>>(wk, wkb, NW);
  cvt_f32_bf16<<<NW / 256, 256, 0, stream>>>(wv, wvb, NW);
  cvt_f32_bf16<<<NW / 256, 256, 0, stream>>>(wo, wob, NW);

  dim3 gg(DMODEL / 64, MROWS / 128);          // 16 x 32 WGs, 256 thr
  gemm_bf16<0><<<gg, 256, 0, stream>>>(xb, wqb, nullptr, qh);
  gemm_bf16<0><<<gg, 256, 0, stream>>>(xb, wkb, nullptr, kh);
  gemm_bf16<1><<<gg, 256, 0, stream>>>(xb, wvb, nullptr, vtb);

  const int NROWS = 2 * NHEAD * S_LEN;        // 65536 head-rows
  rf_feat<1><<<NROWS / 8, 256, 0, stream>>>(qh, frq, phq, sgq, tmp, qfb);
  rf_feat<0><<<NROWS / 8, 256, 0, stream>>>(kh, frk, phk, sgk, tmp, kfb);

  dim3 ga(S_LEN / 128, 2 * NHEAD);            // 16 x 32 WGs, 128 thr
  rfattn<<<ga, 128, 0, stream>>>(qfb, kfb, vtb, oab);

  gemm_bf16<2><<<gg, 256, 0, stream>>>(oab, wob, (float*)d_out, nullptr);
}
